// ContrastiveNet_77549929496845
// MI455X (gfx1250) — compile-verified, matrix operand order checked by reading, not run on hardware
//
#include <hip/hip_runtime.h>
#include <hip/hip_bf16.h>

typedef __attribute__((ext_vector_type(16))) _Float16 v16h;
typedef __attribute__((ext_vector_type(8)))  _Float16 v8h;
typedef __attribute__((ext_vector_type(8)))  float    v8f;

#define B_ROWS 8192
#define D_DIM  512
#define PD_DIM 128
#define NUM_NEG 10
#define INV_TEMP 2.0f   // 1 / 0.5
#define EPS_F 1e-8f

// ---------------------------------------------------------------------------
// WMMA helper (CDNA5 V_WMMA_F32_16X16X32_F16, wave32)
// ---------------------------------------------------------------------------
__device__ __forceinline__ v8f wmma16(v16h a, v16h b, v8f c) {
  // (neg_a, A, neg_b, B, c_mod, C, reuse_a, reuse_b)
  return __builtin_amdgcn_wmma_f32_16x16x32_f16(false, a, false, b, (short)0, c,
                                                false, false);
}

// A/B operand fragment layout (ISA 7.12.2, 16-bit A 16x32):
//  lane&15 selects the row (A) / column (B); lane>>4 selects K-half:
//  elems 0..7  <- K = kb + koff + 0..7
//  elems 8..15 <- K = kb + koff + 16..23          (koff = (lane>>4)*8)
// B mirrored (column n of B in lane n, same K striping); both GEMMs are
// X @ W^T, so B columns are weight ROWS -> contiguous along K in memory.
__device__ __forceinline__ v16h frag_from_f16(const _Float16* __restrict__ p) {
  v8h lo = *(const v8h*)(p);        // K chunk +0..7   (16B load)
  v8h hi = *(const v8h*)(p + 16);   // K chunk +16..23 (16B load)
  return __builtin_shufflevector(lo, hi, 0,1,2,3,4,5,6,7,8,9,10,11,12,13,14,15);
}

// ---------------------------------------------------------------------------
// Elementwise f32 -> f16 conversion: 8 elements / thread, exact grids.
// ---------------------------------------------------------------------------
__global__ void cvt_f16_kernel(const float* __restrict__ in,
                               _Float16* __restrict__ out) {
  const size_t i = ((size_t)blockIdx.x * blockDim.x + threadIdx.x) * 8;
  float4 a = *(const float4*)(in + i);
  float4 b = *(const float4*)(in + i + 4);
  v8h o;
  o[0]=(_Float16)a.x; o[1]=(_Float16)a.y; o[2]=(_Float16)a.z; o[3]=(_Float16)a.w;
  o[4]=(_Float16)b.x; o[5]=(_Float16)b.y; o[6]=(_Float16)b.z; o[7]=(_Float16)b.w;
  *(v8h*)(out + i) = o;
}

// ---------------------------------------------------------------------------
// GEMM1: t[8192x512] = relu(xh[8192x512] @ w1h[512x512]^T), f16 out.
// Block = 256 threads = 8 waves; wave computes 32x64 (8 WMMA / K-step).
// Grid (8192/256, 512/64) = (32, 8). Exact cover -> EXEC all-ones.
// ---------------------------------------------------------------------------
__global__ void gemm1_relu_kernel(const _Float16* __restrict__ xh,
                                  const _Float16* __restrict__ w1h,
                                  _Float16* __restrict__ t) {
  const int K = D_DIM, N = D_DIM;
  const int wid  = threadIdx.x >> 5;
  const int lane = threadIdx.x & 31;
  const int mbase = blockIdx.x * 256 + wid * 32;
  const int nbase = blockIdx.y * 64;
  const int r    = lane & 15;
  const int koff = (lane >> 4) << 3;

  v8f acc[2][4] = {};
  const _Float16* pa = xh  + (size_t)(mbase + r) * K + koff;
  const _Float16* pb = w1h + (size_t)(nbase + r) * K + koff;

#pragma unroll 2
  for (int kb = 0; kb < K; kb += 32) {
    // prefetch next K-block of the streamed A rows (global_prefetch_b8)
    __builtin_prefetch(pa + kb + 32, 0, 1);
    __builtin_prefetch(pa + (size_t)16 * K + kb + 32, 0, 1);

    v16h a0 = frag_from_f16(pa + kb);
    v16h a1 = frag_from_f16(pa + (size_t)16 * K + kb);
    v16h b0 = frag_from_f16(pb + kb);
    v16h b1 = frag_from_f16(pb + (size_t)16 * K + kb);
    v16h b2 = frag_from_f16(pb + (size_t)32 * K + kb);
    v16h b3 = frag_from_f16(pb + (size_t)48 * K + kb);
    acc[0][0] = wmma16(a0, b0, acc[0][0]);
    acc[0][1] = wmma16(a0, b1, acc[0][1]);
    acc[0][2] = wmma16(a0, b2, acc[0][2]);
    acc[0][3] = wmma16(a0, b3, acc[0][3]);
    acc[1][0] = wmma16(a1, b0, acc[1][0]);
    acc[1][1] = wmma16(a1, b1, acc[1][1]);
    acc[1][2] = wmma16(a1, b2, acc[1][2]);
    acc[1][3] = wmma16(a1, b3, acc[1][3]);
  }

  // C/D layout: lane&15 = column, VGPR i = row i (+8 for lanes 16..31)
  const int col = lane & 15;
  const int mo  = (lane >> 4) << 3;
#pragma unroll
  for (int mi = 0; mi < 2; mi++) {
    _Float16* to = t + (size_t)(mbase + mi * 16 + mo) * N + nbase + col;
#pragma unroll
    for (int i = 0; i < 8; i++) {
#pragma unroll
      for (int ni = 0; ni < 4; ni++)
        to[(size_t)i * N + ni * 16] = (_Float16)fmaxf(acc[mi][ni][i], 0.0f);
    }
  }
}

// ---------------------------------------------------------------------------
// GEMM2: h[8192x128] = t[8192x512](f16) @ w2h[128x512]^T, f32 out.
// Grid (32, 2), block 256, wave computes 32x64.
// ---------------------------------------------------------------------------
__global__ void gemm2_kernel(const _Float16* __restrict__ t,
                             const _Float16* __restrict__ w2h,
                             float* __restrict__ h) {
  const int K = D_DIM, N = PD_DIM;
  const int wid  = threadIdx.x >> 5;
  const int lane = threadIdx.x & 31;
  const int mbase = blockIdx.x * 256 + wid * 32;
  const int nbase = blockIdx.y * 64;
  const int r    = lane & 15;
  const int koff = (lane >> 4) << 3;

  v8f acc[2][4] = {};
  const _Float16* pa = t   + (size_t)(mbase + r) * K + koff;
  const _Float16* pb = w2h + (size_t)(nbase + r) * K + koff;

#pragma unroll 2
  for (int kb = 0; kb < K; kb += 32) {
    __builtin_prefetch(pa + kb + 32, 0, 1);
    __builtin_prefetch(pa + (size_t)16 * K + kb + 32, 0, 1);

    v16h a0 = frag_from_f16(pa + kb);
    v16h a1 = frag_from_f16(pa + (size_t)16 * K + kb);
    v16h b0 = frag_from_f16(pb + kb);
    v16h b1 = frag_from_f16(pb + (size_t)16 * K + kb);
    v16h b2 = frag_from_f16(pb + (size_t)32 * K + kb);
    v16h b3 = frag_from_f16(pb + (size_t)48 * K + kb);
    acc[0][0] = wmma16(a0, b0, acc[0][0]);
    acc[0][1] = wmma16(a0, b1, acc[0][1]);
    acc[0][2] = wmma16(a0, b2, acc[0][2]);
    acc[0][3] = wmma16(a0, b3, acc[0][3]);
    acc[1][0] = wmma16(a1, b0, acc[1][0]);
    acc[1][1] = wmma16(a1, b1, acc[1][1]);
    acc[1][2] = wmma16(a1, b2, acc[1][2]);
    acc[1][3] = wmma16(a1, b3, acc[1][3]);
  }

  const int col = lane & 15;
  const int mo  = (lane >> 4) << 3;
#pragma unroll
  for (int mi = 0; mi < 2; mi++) {
    float* ho = h + (size_t)(mbase + mi * 16 + mo) * N + nbase + col;
#pragma unroll
    for (int i = 0; i < 8; i++) {
#pragma unroll
      for (int ni = 0; ni < 4; ni++)
        ho[(size_t)i * N + ni * 16] = acc[mi][ni][i];
    }
  }
}

// ---------------------------------------------------------------------------
// Row norms: one wave (32 lanes) per row of h (128 floats -> float4/lane).
// ---------------------------------------------------------------------------
__global__ void row_norms_kernel(const float* __restrict__ h,
                                 float* __restrict__ norms) {
  const int gw   = (int)((blockIdx.x * blockDim.x + threadIdx.x) >> 5);
  const int lane = threadIdx.x & 31;
  float4 v = *(const float4*)(h + (size_t)gw * PD_DIM + lane * 4);
  float s = v.x * v.x + v.y * v.y + v.z * v.z + v.w * v.w;
#pragma unroll
  for (int off = 16; off > 0; off >>= 1) s += __shfl_xor(s, off, 32);
  if (lane == 0) norms[gw] = sqrtf(s);
}

__global__ void zero_out_kernel(float* __restrict__ out) { out[0] = 0.0f; }

// ---------------------------------------------------------------------------
// Pair loss: one wave per anchor (grid-strided). h rows are L2-resident (4MB).
// 11 gathered 128-dim dots, xor-reduce -> lane-uniform logits, logsumexp,
// block-level LDS reduce, one atomicAdd per block (2048 atomics total).
// ---------------------------------------------------------------------------
__global__ void pair_loss_kernel(const float* __restrict__ h,
                                 const float* __restrict__ norms,
                                 const int* __restrict__ anchors,
                                 const int* __restrict__ positives,
                                 const int* __restrict__ negidx,
                                 int npairs, float inv_n,
                                 float* __restrict__ out) {
  __shared__ float partial[8];
  const int wid  = threadIdx.x >> 5;
  const int lane = threadIdx.x & 31;
  const int gw0    = blockIdx.x * 8 + wid;
  const int stride = gridDim.x * 8;

  float acc = 0.0f;
  for (int i = gw0; i < npairs; i += stride) {
    const int a = anchors[i];
    const float na = norms[a];
    float4 av = *(const float4*)(h + (size_t)a * PD_DIM + lane * 4);

    float logits[1 + NUM_NEG];
#pragma unroll
    for (int j = 0; j < 1 + NUM_NEG; j++) {
      const int p = (j == 0) ? positives[i]
                             : negidx[(size_t)i * NUM_NEG + (j - 1)];
      float4 bv = *(const float4*)(h + (size_t)p * PD_DIM + lane * 4);
      float d = av.x * bv.x + av.y * bv.y + av.z * bv.z + av.w * bv.w;
#pragma unroll
      for (int off = 16; off > 0; off >>= 1) d += __shfl_xor(d, off, 32);
      logits[j] = d / fmaxf(na * norms[p], EPS_F) * INV_TEMP;
    }
    float m = logits[0];
#pragma unroll
    for (int j = 1; j < 1 + NUM_NEG; j++) m = fmaxf(m, logits[j]);
    float s = 0.0f;
#pragma unroll
    for (int j = 0; j < 1 + NUM_NEG; j++) s += __expf(logits[j] - m);
    acc += (m + __logf(s) - logits[0]);   // lse - pos
  }

  if (lane == 0) partial[wid] = acc;
  __syncthreads();
  if (threadIdx.x == 0) {
    float s = 0.0f;
#pragma unroll
    for (int i = 0; i < 8; i++) s += partial[i];
    atomicAdd(out, s * inv_n);
  }
}

// ---------------------------------------------------------------------------
// Launch
// ---------------------------------------------------------------------------
extern "C" void kernel_launch(void* const* d_in, const int* in_sizes, int n_in,
                              void* d_out, int out_size, void* d_ws, size_t ws_size,
                              hipStream_t stream) {
  (void)n_in; (void)out_size; (void)ws_size;
  const float* x  = (const float*)d_in[0];
  const float* w1 = (const float*)d_in[1];
  const float* w2 = (const float*)d_in[2];
  // d_in[3] = targets (unused; pair indices are precomputed)
  const int* anchors   = (const int*)d_in[4];
  const int* positives = (const int*)d_in[5];
  const int* negidx    = (const int*)d_in[6];
  const int npairs = in_sizes[4];

  // Workspace layout (~20.7 MB total)
  char* ws = (char*)d_ws;
  _Float16* xh  = (_Float16*)ws;                                  // 8 MB
  _Float16* w1h = (_Float16*)(ws + (size_t)8388608);              // 512 KB
  _Float16* w2h = (_Float16*)(ws + (size_t)8912896);              // 128 KB
  _Float16* t   = (_Float16*)(ws + (size_t)9043968);              // 8 MB
  float*    h   = (float*)   (ws + (size_t)17432576);             // 4 MB
  float*  norms = (float*)   (ws + (size_t)21626880);             // 32 KB
  float* out = (float*)d_out;

  // f32 -> f16 (8 elems/thread, 256 threads/block, exact grids)
  cvt_f16_kernel<<<(B_ROWS * D_DIM) / 2048, 256, 0, stream>>>(x, xh);
  cvt_f16_kernel<<<(D_DIM * D_DIM) / 2048, 256, 0, stream>>>(w1, w1h);
  cvt_f16_kernel<<<(PD_DIM * D_DIM) / 2048, 256, 0, stream>>>(w2, w2h);

  gemm1_relu_kernel<<<dim3(B_ROWS / 256, D_DIM / 64), 256, 0, stream>>>(xh, w1h, t);
  gemm2_kernel<<<dim3(B_ROWS / 256, PD_DIM / 64), 256, 0, stream>>>(t, w2h, h);
  row_norms_kernel<<<B_ROWS / 8, 256, 0, stream>>>(h, norms);
  zero_out_kernel<<<1, 1, 0, stream>>>(out);
  pair_loss_kernel<<<2048, 256, 0, stream>>>(h, norms, anchors, positives, negidx,
                                             npairs, 1.0f / (float)npairs, out);
}
